// Model_39298950758417
// MI455X (gfx1250) — matrix operand order repaired and checked by reading, not compiled, not run on hardware
//
#include <hip/hip_runtime.h>
#include <hip/hip_bf16.h>
#include <stdint.h>

// ---------------- problem constants ----------------
#define BSZ 2
#define NN  16384
#define DEG 16
#define NEDGE (NN * DEG)      // 262144
#define DD  128
#define SBLK 4

// ---------------- workspace layout (float units) ----------------
#define ACC0_OFF 0                      // [BSZ*NN*64] scatter sum e0 (by sender)
#define ACC1_OFF 2097152                // [BSZ*NN*64] scatter sum e1 (by receiver)
#define CNT0_OFF 4194304                // [BSZ*NN]
#define CNT1_OFF 4227072                // [BSZ*NN]
#define NEMB_OFF 4259840                // [BSZ*NN*128] node_emb
#define WE1P_OFF 8454144                // 192*128 packed words (bf16 pairs)
#define WE2P_OFF 8478720                // 64*128
#define WN1P_OFF 8486912                // 128*128
#define WN2P_OFF 8503296                // 64*128
// total ~ 8.51M floats = ~34 MB of d_ws

typedef __bf16 v16bf __attribute__((ext_vector_type(16)));
typedef float  v8f   __attribute__((ext_vector_type(8)));

union FragBf { v16bf v; uint32_t w[8]; };

static __device__ __forceinline__ uint32_t pk_bf16(float a, float b) {
    // round-to-nearest-even f32 -> bf16, packed (lo = a, hi = b)
    uint32_t ua = __float_as_uint(a), ub = __float_as_uint(b);
    ua = (ua + 0x7FFFu + ((ua >> 16) & 1u)) >> 16;
    ub = (ub + 0x7FFFu + ((ub >> 16) & 1u)) >> 16;
    return (ub << 16) | (ua & 0xFFFFu);
}

static __device__ __forceinline__ float silu(float x) {
    // x * sigmoid(x) with v_rcp_f32 instead of the IEEE divide sequence
    return x * __builtin_amdgcn_rcpf(1.0f + __expf(-x));
}

// A-fragment (16x32 bf16): lane = (m16, half); VGPR v holds K pair starting at:
static __device__ __forceinline__ int a_k0(int v, int half) {
    return 2 * v + 16 * half;
}

// ==================================================================
// Kernel 0: zero accumulators + pack weights to bf16 pair-packed words
// word[kp * 128 + n] = pack(bf16(W[2kp][n]), bf16(W[2kp+1][n]))
// ==================================================================
__global__ __launch_bounds__(256) void prep_kernel(
    const float* __restrict__ We1, const float* __restrict__ We2,
    const float* __restrict__ Wn1, const float* __restrict__ Wn2,
    float* __restrict__ ws)
{
    int tid = blockIdx.x * 256 + threadIdx.x;
    if (tid < NEMB_OFF) ws[tid] = 0.0f;   // acc0, acc1, cnt0, cnt1

    if (tid < 57344) {
        int t = tid;
        const float* W; uint32_t* dst;
        if (t < 24576)            { W = We1; dst = (uint32_t*)(ws + WE1P_OFF); }
        else if (t < 32768)       { t -= 24576; W = We2; dst = (uint32_t*)(ws + WE2P_OFF); }
        else if (t < 49152)       { t -= 32768; W = Wn1; dst = (uint32_t*)(ws + WN1P_OFF); }
        else                      { t -= 49152; W = Wn2; dst = (uint32_t*)(ws + WN2P_OFF); }
        int kp = t >> 7, n = t & 127;
        dst[t] = pk_bf16(W[(2 * kp) * DD + n], W[(2 * kp + 1) * DD + n]);
    }
}

// ==================================================================
// Kernel 1: edge MLP + scatter-add.  1 wave handles 16 edges.
// block = 128 threads = 4 waves; grid = BSZ*NEDGE/64 = 8192
// ==================================================================
__global__ __launch_bounds__(128) void edge_mlp_kernel(
    const float* __restrict__ V, const float* __restrict__ E,
    const int* __restrict__ edges,
    const float* __restrict__ be1, const float* __restrict__ be2,
    float* __restrict__ ws)
{
    __shared__ int s_idx[4][16][2];
    __shared__ __align__(16) uint32_t s_x[4][3072];  // 16 rows x 192 words (bf16 pairs)

    const int lane = threadIdx.x & 31;
    const int wv   = threadIdx.x >> 5;
    const int m16  = lane & 15;
    const int half = lane >> 4;

    const int tile  = blockIdx.x * 4 + wv;
    const int ebase = tile * 16;
    const int b     = ebase / NEDGE;
    const int e0    = ebase - b * NEDGE;

    // ---- gather edge endpoint indices (lanes 0-15 -> sender, 16-31 -> receiver)
    s_idx[wv][m16][half] = edges[((size_t)(b * NEDGE + e0 + m16) << 1) + half];
    asm volatile("s_wait_dscnt 0x0" ::: "memory");

    // ---- degree counts
    if (half == 0) atomicAdd(ws + CNT0_OFF + b * NN + s_idx[wv][m16][0], 1.0f);
    else           atomicAdd(ws + CNT1_OFF + b * NN + s_idx[wv][m16][1], 1.0f);

    // ---- stage x = [V[snd] | V[rcv] | E[e]] as bf16 pairs (16 x 192 words)
    uint32_t* xw = s_x[wv];
    for (int it = 0; it < 96; ++it) {
        int flat = it * 32 + lane;
        int row  = flat / 192;
        int kp   = flat - row * 192;
        int col  = kp * 2;
        const float* src;
        if (col < 128)      src = V + (((size_t)b * NN + s_idx[wv][row][0]) << 7) + col;
        else if (col < 256) src = V + (((size_t)b * NN + s_idx[wv][row][1]) << 7) + (col - 128);
        else                src = E + (((size_t)b * NEDGE + e0 + row) << 7) + (col - 256);
        xw[row * 192 + kp] = pk_bf16(src[0], src[1]);
    }
    asm volatile("s_wait_dscnt 0x0" ::: "memory");

    // ---- GEMM1: x(16x384) @ We1(384x128), fp32 accum
    const uint32_t* We1p = (const uint32_t*)(ws + WE1P_OFF);
    v8f acc[8];
    const v8f vzero = {0.f,0.f,0.f,0.f,0.f,0.f,0.f,0.f};
#pragma unroll
    for (int nt = 0; nt < 8; ++nt) acc[nt] = vzero;

    for (int kt = 0; kt < 12; ++kt) {
        FragBf A;
#pragma unroll
        for (int v = 0; v < 8; ++v)
            A.w[v] = xw[m16 * 192 + kt * 16 + (a_k0(v, half) >> 1)];
#pragma unroll
        for (int nt = 0; nt < 8; ++nt) {
            FragBf B;
#pragma unroll
            for (int v = 0; v < 8; ++v)
                B.w[v] = We1p[(size_t)(kt * 16 + v + 8 * half) * 128 + nt * 16 + m16];
            acc[nt] = __builtin_amdgcn_wmma_f32_16x16x32_bf16(
                false, A.v, false, B.v, (short)0, acc[nt], false, false);
        }
    }

    // ---- bias + SiLU, hidden -> LDS (reuse x region; 16x128 f32)
    float* h = (float*)xw;
#pragma unroll
    for (int nt = 0; nt < 8; ++nt) {
        float bias = be1[nt * 16 + m16];
#pragma unroll
        for (int r = 0; r < 8; ++r)
            h[(r + 8 * half) * 128 + nt * 16 + m16] = silu(acc[nt][r] + bias);
    }
    asm volatile("s_wait_dscnt 0x0" ::: "memory");

    // ---- GEMM2: h(16x128) @ We2(128x128), then split/scatter
    const uint32_t* We2p = (const uint32_t*)(ws + WE2P_OFF);
    float* a0 = ws + ACC0_OFF;
    float* a1 = ws + ACC1_OFF;
#pragma unroll
    for (int nt = 0; nt < 8; ++nt) {
        v8f c = vzero;
#pragma unroll
        for (int kt = 0; kt < 4; ++kt) {
            FragBf A2, B2;
#pragma unroll
            for (int v = 0; v < 8; ++v) {
                int k = kt * 32 + a_k0(v, half);
                float2 hv = *(const float2*)(h + m16 * 128 + k);
                A2.w[v] = pk_bf16(hv.x, hv.y);
                B2.w[v] = We2p[(size_t)(kt * 16 + v + 8 * half) * 128 + nt * 16 + m16];
            }
            c = __builtin_amdgcn_wmma_f32_16x16x32_bf16(
                false, A2.v, false, B2.v, (short)0, c, false, false);
        }
        float bias = be2[nt * 16 + m16];
#pragma unroll
        for (int r = 0; r < 8; ++r) {
            float val = c[r] + bias;
            int mm = r + 8 * half;
            if (nt < 4)
                atomicAdd(a0 + (((size_t)b * NN + s_idx[wv][mm][0]) << 6) + nt * 16 + m16, val);
            else
                atomicAdd(a1 + (((size_t)b * NN + s_idx[wv][mm][1]) << 6) + (nt - 4) * 16 + m16, val);
        }
    }
}

// ==================================================================
// Kernel 2: node MLP.  1 wave = 16 nodes. grid = BSZ*NN/64 = 512
// ==================================================================
__global__ __launch_bounds__(128) void node_mlp_kernel(
    const float* __restrict__ V,
    const float* __restrict__ bn1, const float* __restrict__ bn2,
    float* __restrict__ ws)
{
    __shared__ __align__(16) uint32_t s_x[4][2048];  // 16 rows x 128 words

    const int lane = threadIdx.x & 31;
    const int wv   = threadIdx.x >> 5;
    const int m16  = lane & 15;
    const int half = lane >> 4;

    const int tile = blockIdx.x * 4 + wv;
    const int base = tile * 16;
    const int b    = base / NN;
    const int n0   = base - b * NN;

    const float* a0   = ws + ACC0_OFF;
    const float* a1   = ws + ACC1_OFF;
    const float* cnt0 = ws + CNT0_OFF;
    const float* cnt1 = ws + CNT1_OFF;

    // ---- stage x = [V | m0 | m1] as bf16 pairs (mean fused into staging)
    uint32_t* xw = s_x[wv];
    for (int it = 0; it < 64; ++it) {
        int flat = it * 32 + lane;
        int row  = flat >> 7;
        int j    = flat & 127;
        int col  = 2 * j;
        size_t node = (size_t)b * NN + n0 + row;
        float f0, f1;
        if (col < 128) {
            const float* s = V + (node << 7) + col;
            f0 = s[0]; f1 = s[1];
        } else if (col < 192) {
            float cinv = __builtin_amdgcn_rcpf(fmaxf(cnt0[node], 1.0f));
            const float* s = a0 + (node << 6) + (col - 128);
            f0 = s[0] * cinv; f1 = s[1] * cinv;
        } else {
            float cinv = __builtin_amdgcn_rcpf(fmaxf(cnt1[node], 1.0f));
            const float* s = a1 + (node << 6) + (col - 192);
            f0 = s[0] * cinv; f1 = s[1] * cinv;
        }
        xw[row * 128 + j] = pk_bf16(f0, f1);
    }
    asm volatile("s_wait_dscnt 0x0" ::: "memory");

    // ---- GEMM1: x(16x256) @ Wn1(256x128)
    const uint32_t* Wn1p = (const uint32_t*)(ws + WN1P_OFF);
    v8f acc[8];
    const v8f vzero = {0.f,0.f,0.f,0.f,0.f,0.f,0.f,0.f};
#pragma unroll
    for (int nt = 0; nt < 8; ++nt) acc[nt] = vzero;

    for (int kt = 0; kt < 8; ++kt) {
        FragBf A;
#pragma unroll
        for (int v = 0; v < 8; ++v)
            A.w[v] = xw[m16 * 128 + kt * 16 + (a_k0(v, half) >> 1)];
#pragma unroll
        for (int nt = 0; nt < 8; ++nt) {
            FragBf B;
#pragma unroll
            for (int v = 0; v < 8; ++v)
                B.w[v] = Wn1p[(size_t)(kt * 16 + v + 8 * half) * 128 + nt * 16 + m16];
            acc[nt] = __builtin_amdgcn_wmma_f32_16x16x32_bf16(
                false, A.v, false, B.v, (short)0, acc[nt], false, false);
        }
    }

    // ---- bias + SiLU, hidden -> LDS (reuse x region exactly)
    float* h = (float*)xw;
#pragma unroll
    for (int nt = 0; nt < 8; ++nt) {
        float bias = bn1[nt * 16 + m16];
#pragma unroll
        for (int r = 0; r < 8; ++r)
            h[(r + 8 * half) * 128 + nt * 16 + m16] = silu(acc[nt][r] + bias);
    }
    asm volatile("s_wait_dscnt 0x0" ::: "memory");

    // ---- GEMM2: h(16x128) @ Wn2(128x128) -> node_emb
    const uint32_t* Wn2p = (const uint32_t*)(ws + WN2P_OFF);
    float* nemb = ws + NEMB_OFF;
#pragma unroll
    for (int nt = 0; nt < 8; ++nt) {
        v8f c = vzero;
#pragma unroll
        for (int kt = 0; kt < 4; ++kt) {
            FragBf A2, B2;
#pragma unroll
            for (int v = 0; v < 8; ++v) {
                int k = kt * 32 + a_k0(v, half);
                float2 hv = *(const float2*)(h + m16 * 128 + k);
                A2.w[v] = pk_bf16(hv.x, hv.y);
                B2.w[v] = Wn2p[(size_t)(kt * 16 + v + 8 * half) * 128 + nt * 16 + m16];
            }
            c = __builtin_amdgcn_wmma_f32_16x16x32_bf16(
                false, A2.v, false, B2.v, (short)0, c, false, false);
        }
        float bias = bn2[nt * 16 + m16];
#pragma unroll
        for (int r = 0; r < 8; ++r) {
            int mm = r + 8 * half;
            nemb[(((size_t)b * NN + n0 + mm) << 7) + nt * 16 + m16] = c[r] + bias;
        }
    }
}

// ==================================================================
// Kernel 3: depth-wise block attention.  1 wave per node (wave32:
// each lane owns 4 of the 128 dims).  grid = BSZ*NN/8 = 4096, block 256.
// logit[s] = rsqrt(mean(x^2)+eps) * sum_d (w_attn[d]*rms_w[d]) * x[d]
// ==================================================================
__global__ __launch_bounds__(256) void attn_kernel(
    const float* __restrict__ blocks,
    const float* __restrict__ w_attn, const float* __restrict__ rms_w,
    const float* __restrict__ nemb, float* __restrict__ out)
{
    const int lane = threadIdx.x & 31;
    const int wv   = threadIdx.x >> 5;
    const int nf   = blockIdx.x * 8 + wv;
    const int b    = nf / NN;
    const int n    = nf - b * NN;
    const int d0   = lane * 4;

    float4 wa = *(const float4*)(w_attn + d0);
    float4 rw = *(const float4*)(rms_w + d0);
    float w0 = wa.x * rw.x, w1 = wa.y * rw.y, w2 = wa.z * rw.z, w3 = wa.w * rw.w;

    float vs[SBLK + 1][4];
    float logit[SBLK + 1];
#pragma unroll
    for (int s = 0; s <= SBLK; ++s) {
        const float* src = (s < SBLK)
            ? blocks + (((((size_t)s * BSZ + b) * NN) + n) << 7) + d0
            : nemb + (((size_t)b * NN + n) << 7) + d0;
        float4 v = *(const float4*)src;
        vs[s][0] = v.x; vs[s][1] = v.y; vs[s][2] = v.z; vs[s][3] = v.w;
        float ss = v.x * v.x + v.y * v.y + v.z * v.z + v.w * v.w;
        float dt = w0 * v.x + w1 * v.y + w2 * v.z + w3 * v.w;
#pragma unroll
        for (int off = 16; off > 0; off >>= 1) {
            ss += __shfl_xor(ss, off, 32);
            dt += __shfl_xor(dt, off, 32);
        }
        logit[s] = dt * rsqrtf(ss * (1.0f / 128.0f) + 1e-6f);
    }

    float mx = logit[0];
#pragma unroll
    for (int s = 1; s <= SBLK; ++s) mx = fmaxf(mx, logit[s]);
    float esum = 0.0f, al[SBLK + 1];
#pragma unroll
    for (int s = 0; s <= SBLK; ++s) { al[s] = __expf(logit[s] - mx); esum += al[s]; }
    float inv = __builtin_amdgcn_rcpf(esum);

    float o0 = 0, o1 = 0, o2 = 0, o3 = 0;
#pragma unroll
    for (int s = 0; s <= SBLK; ++s) {
        float a = al[s] * inv;
        o0 += a * vs[s][0]; o1 += a * vs[s][1]; o2 += a * vs[s][2]; o3 += a * vs[s][3];
    }
    float4 o = {o0, o1, o2, o3};
    *(float4*)(out + (((size_t)b * NN + n) << 7) + d0) = o;
}

// ==================================================================
extern "C" void kernel_launch(void* const* d_in, const int* in_sizes, int n_in,
                              void* d_out, int out_size, void* d_ws, size_t ws_size,
                              hipStream_t stream) {
    const float* V      = (const float*)d_in[0];
    const float* E      = (const float*)d_in[1];
    const float* blocks = (const float*)d_in[2];
    const float* We1    = (const float*)d_in[3];
    const float* be1    = (const float*)d_in[4];
    const float* We2    = (const float*)d_in[5];
    const float* be2    = (const float*)d_in[6];
    const float* Wn1    = (const float*)d_in[7];
    const float* bn1    = (const float*)d_in[8];
    const float* Wn2    = (const float*)d_in[9];
    const float* bn2    = (const float*)d_in[10];
    const float* w_attn = (const float*)d_in[11];
    const float* rms_w  = (const float*)d_in[12];
    const int*   edges  = (const int*)d_in[13];

    float* ws  = (float*)d_ws;
    float* out = (float*)d_out;

    prep_kernel<<<16640, 256, 0, stream>>>(We1, We2, Wn1, Wn2, ws);
    edge_mlp_kernel<<<(BSZ * NEDGE) / 64, 128, 0, stream>>>(V, E, edges, be1, be2, ws);
    node_mlp_kernel<<<(BSZ * NN) / 64, 128, 0, stream>>>(V, bn1, bn2, ws);
    attn_kernel<<<(BSZ * NN) / 8, 256, 0, stream>>>(blocks, w_attn, rms_w,
                                                    ws + NEMB_OFF, out);
}
